// UpsampleConvLayerIncr_17575006175564
// MI455X (gfx1250) — matrix operand order repaired and checked by reading, not compiled, not run on hardware
//
#include <hip/hip_runtime.h>
#include <hip/hip_bf16.h>

typedef __bf16 bf16_t;
typedef __attribute__((ext_vector_type(16))) __bf16 v16bf;
typedef __attribute__((ext_vector_type(8)))  float  v8f;

union FragU { v16bf v; uint4 u[2]; };

// A fragment (16-bit A layout, ISA 7.12.2): per lane two 8-elem runs at +0 and +16
__device__ __forceinline__ v16bf load_a_frag(const bf16_t* p) {
  FragU f;
  f.u[0] = *(const uint4*)(p);
  f.u[1] = *(const uint4*)(p + 16);
  return f.v;
}
// B fragment: per lane two 8-elem runs at +0 and +16 (same K order as A)
__device__ __forceinline__ v16bf load_b_frag(const bf16_t* p) {
  FragU f;
  f.u[0] = *(const uint4*)(p);
  f.u[1] = *(const uint4*)(p + 16);
  return f.v;
}

// ---------------------------------------------------------------------------
// Pass 1: bilinear x2 upsample (half-pixel centers) + f32->bf16 pack, NHWC
// with 2-pixel zero halo:  up[b][y=0..131][x=0..131][c=0..255]
// ---------------------------------------------------------------------------
__global__ void upsample_pack(const float* __restrict__ x, bf16_t* __restrict__ up) {
  const int c  = threadIdx.x;          // 0..255 channels (coalesced bf16 writes)
  const int b  = blockIdx.y;           // 0..7
  const int y  = blockIdx.x / 132;
  const int xx = blockIdx.x % 132;
  const size_t oidx = ((((size_t)b * 132 + y) * 132) + xx) * 256 + c;
  float val = 0.0f;
  if (y >= 2 && y < 130 && xx >= 2 && xx < 130) {
    const int oy = y - 2, ox = xx - 2;
    const int ky = oy >> 1, kx = ox >> 1;
    int iy0, iy1, ix0, ix1; float wy0, wy1, wx0, wx1;
    if (oy & 1) { iy0 = ky;            iy1 = min(ky + 1, 63); wy0 = 0.75f; wy1 = 0.25f; }
    else        { iy0 = max(ky - 1, 0); iy1 = ky;             wy0 = 0.25f; wy1 = 0.75f; }
    if (ox & 1) { ix0 = kx;            ix1 = min(kx + 1, 63); wx0 = 0.75f; wx1 = 0.25f; }
    else        { ix0 = max(kx - 1, 0); ix1 = kx;             wx0 = 0.25f; wx1 = 0.75f; }
    const float* xb = x + (((size_t)b * 256 + c) * 64) * 64;
    const float v00 = xb[iy0 * 64 + ix0], v01 = xb[iy0 * 64 + ix1];
    const float v10 = xb[iy1 * 64 + ix0], v11 = xb[iy1 * 64 + ix1];
    val = wy0 * (wx0 * v00 + wx1 * v01) + wy1 * (wx0 * v10 + wx1 * v11);
  }
  up[oidx] = (bf16_t)val;
}

// ---------------------------------------------------------------------------
// Pass 2: weight pack OIHW f32 -> [tap][o][i] bf16 (i contiguous)
// ---------------------------------------------------------------------------
__global__ void pack_weights(const float* __restrict__ W, bf16_t* __restrict__ Wp) {
  const int idx = blockIdx.x * 256 + threadIdx.x;   // < 25*128*256 = 819200
  const int i   = idx & 255;
  const int o   = (idx >> 8) & 127;
  const int tap = idx >> 15;
  Wp[idx] = (bf16_t)W[((size_t)o * 256 + i) * 25 + tap];
}

__global__ void zero_stats(float* __restrict__ stats) {
  stats[threadIdx.x] = 0.0f;   // 256 threads: sums + sumsqs
}

// ---------------------------------------------------------------------------
// Pass 3: implicit-GEMM conv via v_wmma_f32_16x16x32_bf16.
// grid = (1024 rows [b*128+oy], 2 channel groups of 64), block = 256 (8 waves).
// Wave tile: 32 pixels (M) x 32 channels (N) -> 4 accumulators; A and B
// fragments each reused twice => 2 b128 loads per WMMA (vs 3 for 16x32).
// waves 0-3: ox = {0,32,64,96}, channels nb..nb+31
// waves 4-7: same ox,          channels nb+32..nb+63
// K loop: 25 taps x 8 chunks of 32 input channels = 800 WMMA / wave.
// Stores pre-BN conv output to d_out (NCHW) and accumulates per-channel stats.
// ---------------------------------------------------------------------------
__global__ void conv_wmma_bf16(const bf16_t* __restrict__ up,
                               const bf16_t* __restrict__ Wp,
                               float* __restrict__ out,
                               float* __restrict__ stats) {
  __shared__ float sm[128];            // [0:64) sums, [64:128) sumsqs (block-local chans)
  const int tid  = threadIdx.x;
  const int wave = tid >> 5;
  const int lane = tid & 31;
  if (tid < 128) sm[tid] = 0.0f;
  __syncthreads();

  const int row   = blockIdx.x;        // b*128 + oy
  const int b     = row >> 7;
  const int oy    = row & 127;
  const int grp   = wave >> 2;                          // 0..1 (32-channel subgroup)
  const int nbase = (blockIdx.y << 6) + (grp << 5);     // 32-channel base
  const int ox0   = (wave & 3) << 5;                    // 0,32,64,96
  const int m     = lane & 15;
  const int aklo  = (lane < 16) ? 0 : 8;    // A: runs {0-7,16-23} / {8-15,24-31}
  const int bklo  = (lane < 16) ? 0 : 8;    // B: runs {0-7,16-23} / {8-15,24-31}

  const bf16_t* wbase = Wp + ((size_t)(nbase + m)) * 256 + bklo;

  v8f acc00 = {}, acc01 = {};          // M tile 0 x channels {c0, c0+16}
  v8f acc10 = {}, acc11 = {};          // M tile 1 x channels {c0, c0+16}

  for (int ky = 0; ky < 5; ++ky) {
    const bf16_t* arow0 = up + ((((size_t)b * 132) + oy + ky) * 132 + ox0 + m) * 256 + aklo;
    const bf16_t* arow1 = arow0 + 16 * 256;
    __builtin_prefetch(arow0 + 132 * 256, 0, 0);   // next input row -> global_prefetch
    for (int kx = 0; kx < 5; ++kx) {
      const bf16_t* aptr0 = arow0 + (size_t)kx * 256;
      const bf16_t* aptr1 = arow1 + (size_t)kx * 256;
      const bf16_t* bptr  = wbase + (size_t)(ky * 5 + kx) * 32768;  // 128*256 per tap
#pragma unroll
      for (int cc = 0; cc < 8; ++cc) {
        v16bf a0 = load_a_frag(aptr0 + cc * 32);
        v16bf a1 = load_a_frag(aptr1 + cc * 32);
        v16bf b0 = load_b_frag(bptr + cc * 32);
        v16bf b1 = load_b_frag(bptr + 4096 + cc * 32);   // channels +16
        acc00 = __builtin_amdgcn_wmma_f32_16x16x32_bf16(false, a0, false, b0,
                                                        (short)0, acc00, false, false);
        acc01 = __builtin_amdgcn_wmma_f32_16x16x32_bf16(false, a0, false, b1,
                                                        (short)0, acc01, false, false);
        acc10 = __builtin_amdgcn_wmma_f32_16x16x32_bf16(false, a1, false, b0,
                                                        (short)0, acc10, false, false);
        acc11 = __builtin_amdgcn_wmma_f32_16x16x32_bf16(false, a1, false, b1,
                                                        (short)0, acc11, false, false);
      }
    }
  }

  // ---- store pre-BN conv output (NCHW). Lane holds 8 contiguous ox per acc.
  const int c0  = nbase + m;
  const int off = (lane < 16) ? 0 : 8;
  float* p0 = out + (((size_t)b * 128 + c0) * 128 + oy) * 128 + ox0 + off;       // acc00/acc10
  float* p1 = out + (((size_t)b * 128 + c0 + 16) * 128 + oy) * 128 + ox0 + off;  // acc01/acc11
  *(float4*)(p0)          = make_float4(acc00[0], acc00[1], acc00[2], acc00[3]);
  *(float4*)(p0 + 4)      = make_float4(acc00[4], acc00[5], acc00[6], acc00[7]);
  *(float4*)(p0 + 16)     = make_float4(acc10[0], acc10[1], acc10[2], acc10[3]);
  *(float4*)(p0 + 20)     = make_float4(acc10[4], acc10[5], acc10[6], acc10[7]);
  *(float4*)(p1)          = make_float4(acc01[0], acc01[1], acc01[2], acc01[3]);
  *(float4*)(p1 + 4)      = make_float4(acc01[4], acc01[5], acc01[6], acc01[7]);
  *(float4*)(p1 + 16)     = make_float4(acc11[0], acc11[1], acc11[2], acc11[3]);
  *(float4*)(p1 + 20)     = make_float4(acc11[4], acc11[5], acc11[6], acc11[7]);

  // ---- per-channel sum / sumsq (lanes L and L+16 share a channel)
  float s0 = 0.f, q0 = 0.f, s1 = 0.f, q1 = 0.f;
#pragma unroll
  for (int v = 0; v < 8; ++v) {
    float a, c;
    a = acc00[v]; c = acc10[v]; s0 += a + c; q0 += a * a + c * c;
    a = acc01[v]; c = acc11[v]; s1 += a + c; q1 += a * a + c * c;
  }
  s0 += __shfl_xor(s0, 16); q0 += __shfl_xor(q0, 16);
  s1 += __shfl_xor(s1, 16); q1 += __shfl_xor(q1, 16);
  if (lane < 16) {
    const int lc = (grp << 5) + m;         // block-local channel 0..63
    atomicAdd(&sm[lc],           s0);
    atomicAdd(&sm[64 + lc],      q0);
    atomicAdd(&sm[lc + 16],      s1);
    atomicAdd(&sm[64 + lc + 16], q1);
  }
  __syncthreads();
  if (tid < 64)        atomicAdd(&stats[(blockIdx.y << 6) + tid],              sm[tid]);
  else if (tid < 128)  atomicAdd(&stats[128 + (blockIdx.y << 6) + (tid - 64)], sm[tid]);
}

// ---------------------------------------------------------------------------
// Pass 4: batch statistics -> scale/shift
// ---------------------------------------------------------------------------
__global__ void bn_finalize(float* __restrict__ stats,
                            const float* __restrict__ gamma,
                            const float* __restrict__ beta) {
  const int c = threadIdx.x;                       // 128 threads
  const float inv_n = 1.0f / 131072.0f;            // 8*128*128 pixels
  const float mean = stats[c] * inv_n;
  const float var  = stats[128 + c] * inv_n - mean * mean;
  const float sc   = gamma[c] * rsqrtf(var + 1e-5f);
  stats[256 + c] = sc;
  stats[384 + c] = beta[c] - mean * sc;
}

// ---------------------------------------------------------------------------
// Pass 5: in-place normalize + ReLU over d_out (float4)
// ---------------------------------------------------------------------------
__global__ void bn_relu(float* __restrict__ out, const float* __restrict__ stats) {
  const int t = blockIdx.x * 256 + threadIdx.x;    // float4 index
  const size_t base = (size_t)t * 4;
  const int c = (int)((base >> 14) & 127);         // plane = 128*128 = 16384
  const float sc = stats[256 + c];
  const float sh = stats[384 + c];
  float4* p = (float4*)(out + base);
  float4 v = *p;
  v.x = fmaxf(v.x * sc + sh, 0.0f);
  v.y = fmaxf(v.y * sc + sh, 0.0f);
  v.z = fmaxf(v.z * sc + sh, 0.0f);
  v.w = fmaxf(v.w * sc + sh, 0.0f);
  *p = v;
}

extern "C" void kernel_launch(void* const* d_in, const int* in_sizes, int n_in,
                              void* d_out, int out_size, void* d_ws, size_t ws_size,
                              hipStream_t stream) {
  const float* x     = (const float*)d_in[0];   // (8,256,64,64)
  const float* W     = (const float*)d_in[1];   // (128,256,5,5)
  const float* gamma = (const float*)d_in[2];   // (128,)
  const float* beta  = (const float*)d_in[3];   // (128,)
  float* out = (float*)d_out;                   // (8,128,128,128)

  char* ws = (char*)d_ws;
  bf16_t* up = (bf16_t*)ws;                                   // 8*132*132*256*2 B
  const size_t upBytes = (size_t)8 * 132 * 132 * 256 * 2;     // 71,368,704
  bf16_t* Wp = (bf16_t*)(ws + upBytes);                       // 25*128*256*2 B
  const size_t wpBytes = (size_t)25 * 128 * 256 * 2;          // 1,638,400
  float* stats = (float*)(ws + upBytes + wpBytes);            // 512 floats

  zero_stats    <<<1, 256, 0, stream>>>(stats);
  upsample_pack <<<dim3(132 * 132, 8), 256, 0, stream>>>(x, up);
  pack_weights  <<<3200, 256, 0, stream>>>(W, Wp);
  conv_wmma_bf16<<<dim3(8 * 128, 2), 256, 0, stream>>>(up, Wp, out, stats);
  bn_finalize   <<<1, 128, 0, stream>>>(stats, gamma, beta);
  bn_relu       <<<16384, 256, 0, stream>>>(out, stats);
}
